// nri_vae_38276748541962
// MI455X (gfx1250) — matrix operand order repaired and checked by reading, not compiled, not run on hardware
//
#include <hip/hip_runtime.h>
#include <hip/hip_bf16.h>

// ---------------------------------------------------------------------------
// NRI-VAE forward for MI455X (gfx1250, wave32, WMMA + async-to-LDS).
//
// Algebraic restructuring (exact): concat(h[dst],h[src]) @ W ==
//   (h @ W_top)[dst] + (h @ W_bot)[src], so every edge MLP collapses to
// node-level GEMMs + per-edge gather-adds EXCEPT the x_skip term of the
// second encoder MLP: a true [E,256]@[256,256] GEMM (19.3 GFLOP).
// That GEMM runs as: 512-thread block = 16 waves; the 16x256 f16 A-tile is
// staged into LDS once with global_load_async_to_lds_b128 (ASYNCcnt path),
// then each wave computes one 16x16 N-tile with v_wmma_f32_16x16x32_f16
// reading A-fragments from LDS -> minimal 75 MB HBM traffic for A, B kept
// L2-resident (128 KB), fused gather/bias/relu epilogue.
// ---------------------------------------------------------------------------

#define JN   384
#define EN   147072
#define HC   256
#define NINC 1024
#define NOUTC 1024
#define BNEPS 1e-5f

typedef __attribute__((ext_vector_type(16))) _Float16 v16h;
typedef __attribute__((ext_vector_type(8)))  _Float16 v8h;
typedef __attribute__((ext_vector_type(8)))  float    v8f;

// ------------------------- WMMA fragment loaders ---------------------------
// A (16x32 f16, MxK): lanes 0-15 hold row M=lane, K = kb+{0..7,16..23};
// lanes 16-31 hold row M=lane-16, K = kb+{8..15,24..31}. Two 16B loads.
__device__ __forceinline__ v16h frag_a(const _Float16* __restrict__ A, int lda,
                                       int m0, int kb, int lane) {
  const int row = lane & 15;
  const int kh  = (lane >> 4) << 3;                 // 0 or 8
  const _Float16* p = A + (size_t)(m0 + row) * lda + kb + kh;
  union { v16h v; v8h h[2]; } u;
  u.h[0] = *(const v8h*)(p);
  u.h[1] = *(const v8h*)(p + 16);
  return u.v;
}
// Same fragment shape out of an LDS-resident 16x256 tile (ds_load_b128 x2).
__device__ __forceinline__ v16h frag_a_lds(const _Float16* tile, int kb, int lane) {
  const int row = lane & 15;
  const int kh  = (lane >> 4) << 3;
  const _Float16* p = tile + row * HC + kb + kh;
  union { v16h v; v8h h[2]; } u;
  u.h[0] = *(const v8h*)(p);
  u.h[1] = *(const v8h*)(p + 16);
  return u.v;
}
// B (32x16 f16, KxN) from pre-transposed Bt[N,K] row-major: lane holds
// column N=lane&15, K = kb + (lane>=16?8:0) + {0..7,16..23}. Two 16B loads.
__device__ __forceinline__ v16h frag_b(const _Float16* __restrict__ Bt, int ldb,
                                       int n0, int kb, int lane) {
  const int col = lane & 15;
  const int ks  = (lane >> 4) << 3;                 // 0 or 8
  const _Float16* p = Bt + (size_t)(n0 + col) * ldb + kb + ks;
  union { v16h v; v8h h[2]; } u;
  u.h[0] = *(const v8h*)(p);
  u.h[1] = *(const v8h*)(p + 16);
  return u.v;
}

// ---------------------------------------------------------------------------
// Generic WMMA GEMM for the small node-level matrices (A is L2-resident):
// C[M,N] = A_f16[M,K] @ Bt_f16[N,K]^T, f32 accumulate, plain f32 store.
// One wave per 16x16 tile; block = 128 threads = 4 N-tiles.
// ---------------------------------------------------------------------------
__global__ void k_gemm_wmma(const _Float16* __restrict__ A,
                            const _Float16* __restrict__ Bt,
                            float* __restrict__ C, int M, int N, int K) {
  const int lane = threadIdx.x & 31;
  const int wave = threadIdx.x >> 5;
  const int n0 = (blockIdx.x * 4 + wave) * 16;
  const int m0 = blockIdx.y * 16;
  if (n0 >= N || m0 >= M) return;      // wave-uniform: EXEC stays all-ones

  v8f acc = {};
  for (int kb = 0; kb < K; kb += 32) {
    if (kb + 32 < K)                   // gfx1250 global_prefetch of next block
      __builtin_prefetch(A + (size_t)(m0 + (lane & 15)) * K + kb + 32, 0, 1);
    v16h a = frag_a(A, K, m0, kb, lane);
    v16h b = frag_b(Bt, K, n0, kb, lane);
    acc = __builtin_amdgcn_wmma_f32_16x16x32_f16(
        false, a, false, b, (short)0, acc, false, false);
  }
  const int n  = lane & 15;
  const int mo = (lane >> 4) << 3;     // C/D: VGPR r -> M=r (+8 for hi lanes)
  #pragma unroll
  for (int r = 0; r < 8; ++r)
    C[(size_t)(m0 + mo + r) * N + n0 + n] = acc[r];
}

// ---------------------------------------------------------------------------
// Big edge GEMM with async-to-LDS A staging + fused epilogue:
//   e2[k,c] = relu( (x_skip @ W2c)[k,c] + P[dst[k],c] + Q[src[k],c] + b[c] )
// grid = E/16 blocks of 512 threads (16 waves); wave w owns N-tile w.
// ---------------------------------------------------------------------------
__global__ void __launch_bounds__(512)
k_gemm_edge(const _Float16* __restrict__ A,   // [E, 256] f16 (x_skip)
            const _Float16* __restrict__ Bt,  // [256, 256] f16 (W2c^T)
            float* __restrict__ C,            // [E, 256] f32 (e2 raw)
            const float* __restrict__ P, const float* __restrict__ Q,
            const float* __restrict__ bias,
            const int* __restrict__ src, const int* __restrict__ dst) {
  __shared__ _Float16 Atile[16 * HC];          // 8 KB
  const int m0   = blockIdx.x * 16;
  const int lane = threadIdx.x & 31;
  const int wave = threadIdx.x >> 5;           // 0..15 -> N-tile
  const int n0   = wave * 16;

  // --- stage A tile: 512 threads x one 16B async copy each ---------------
  {
    const int row = threadIdx.x >> 5;          // 0..15
    const int seg = threadIdx.x & 31;          // 0..31 (16B segments per row)
    const _Float16* gp = A + (size_t)(m0 + row) * HC + seg * 8;
    unsigned long long ga = (unsigned long long)(size_t)gp;
    unsigned lds_off = (unsigned)(size_t)(&Atile[row * HC + seg * 8]);
    asm volatile("global_load_async_to_lds_b128 %0, %1, off"
                 :: "v"(lds_off), "v"(ga) : "memory");
  }
  asm volatile("s_wait_asynccnt 0" ::: "memory");
  __syncthreads();

  // --- 8 WMMA steps over K=256, A from LDS, B from L2-resident weights ---
  v8f acc = {};
  #pragma unroll
  for (int kb = 0; kb < HC; kb += 32) {
    v16h a = frag_a_lds(Atile, kb, lane);
    v16h b = frag_b(Bt, HC, n0, kb, lane);
    acc = __builtin_amdgcn_wmma_f32_16x16x32_f16(
        false, a, false, b, (short)0, acc, false, false);
  }

  // --- fused epilogue -----------------------------------------------------
  const int n  = lane & 15;
  const int mo = (lane >> 4) << 3;
  #pragma unroll
  for (int r = 0; r < 8; ++r) {
    const int row = m0 + mo + r;                 // edge index
    const int col = n0 + n;
    const int s = src[row], d = dst[row];
    float v = acc[r] + P[(size_t)d * HC + col] + Q[(size_t)s * HC + col] + bias[col];
    C[(size_t)row * HC + col] = fmaxf(v, 0.0f);
  }
}

// ------------------------------ elementwise --------------------------------
__global__ void k_fill(float* __restrict__ p, size_t n, float v) {
  for (size_t i = (size_t)blockIdx.x * blockDim.x + threadIdx.x; i < n;
       i += (size_t)gridDim.x * blockDim.x)
    p[i] = v;
}

__global__ void k_cvt_f16(const float* __restrict__ in, _Float16* __restrict__ out,
                          size_t n) {
  for (size_t i = (size_t)blockIdx.x * blockDim.x + threadIdx.x; i < n;
       i += (size_t)gridDim.x * blockDim.x)
    out[i] = (_Float16)in[i];
}

// Wt[n*K + k] = (f16) W[(rowoff + k)*N + n]   (transpose + convert)
__global__ void k_wtrans(const float* __restrict__ W, _Float16* __restrict__ Wt,
                         int K, int N, int rowoff) {
  size_t tot = (size_t)K * N;
  for (size_t i = (size_t)blockIdx.x * blockDim.x + threadIdx.x; i < tot;
       i += (size_t)gridDim.x * blockDim.x) {
    int n = (int)(i / K), k = (int)(i % K);
    Wt[i] = (_Float16)W[(size_t)(rowoff + k) * N + n];
  }
}

// deg[dst[k]] += (mask ? mask[k] : 1)
__global__ void k_indeg(const int* __restrict__ dst, const float* __restrict__ mask,
                        float* __restrict__ deg, int E) {
  int i = blockIdx.x * blockDim.x + threadIdx.x;
  if (i >= E) return;
  float w = mask ? mask[i] : 1.0f;
  if (w != 0.0f) atomicAdd(&deg[dst[i]], w);
}

// agg[dst[k], :] += xw[src[k], :] * ew[k] * rsqrt(deg[s]*deg[d])
__global__ void k_gcn_agg(const float* __restrict__ xw, const int* __restrict__ src,
                          const int* __restrict__ dst, const float* __restrict__ deg,
                          const float* __restrict__ mask, float* __restrict__ agg,
                          int F) {
  const int k = blockIdx.x;
  const float w = mask ? mask[k] : 1.0f;
  if (w == 0.0f) return;
  const int s = src[k], d = dst[k];
  const float nm = w * rsqrtf(deg[s] * deg[d]);
  for (int c = threadIdx.x; c < F; c += blockDim.x)
    atomicAdd(&agg[(size_t)d * F + c], xw[(size_t)s * F + c] * nm);
}

// h16[j,c] = (f16) relu(agg + xw/deg + b)
__global__ void k_gcn_fin_f16(const float* __restrict__ agg, const float* __restrict__ xw,
                              const float* __restrict__ deg, const float* __restrict__ b,
                              _Float16* __restrict__ h, int F) {
  const int j = blockIdx.x;
  const float id = 1.0f / deg[j];
  for (int c = threadIdx.x; c < F; c += blockDim.x)
    h[(size_t)j * F + c] =
        (_Float16)fmaxf(agg[(size_t)j * F + c] + xw[(size_t)j * F + c] * id + b[c], 0.0f);
}

// out[j,c] = agg + xw/deg + b   (no relu; final recon)
__global__ void k_gcn_fin_out(const float* __restrict__ agg, const float* __restrict__ xw,
                              const float* __restrict__ deg, const float* __restrict__ b,
                              float* __restrict__ out, int F) {
  const int j = blockIdx.x;
  const float id = 1.0f / deg[j];
  for (int c = threadIdx.x; c < F; c += blockDim.x)
    out[(size_t)j * F + c] =
        agg[(size_t)j * F + c] + xw[(size_t)j * F + c] * id + b[c];
}

// e[k,c] = relu(P[dst[k],c] + Q[src[k],c] + b[c])
__global__ void k_edge_mlp(const float* __restrict__ P, const float* __restrict__ Q,
                           const float* __restrict__ b, const int* __restrict__ src,
                           const int* __restrict__ dst, float* __restrict__ e) {
  const int k = blockIdx.x, c = threadIdx.x;
  const int s = src[k], d = dst[k];
  e[(size_t)k * HC + c] =
      fmaxf(P[(size_t)d * HC + c] + Q[(size_t)s * HC + c] + b[c], 0.0f);
}

// per-channel weighted sums for batchnorm: S1 += w*x, S2 += w*x*x
__global__ void k_bn_stats(const float* __restrict__ X, const float* __restrict__ mask,
                           float* __restrict__ S1, float* __restrict__ S2, int E) {
  const int c = threadIdx.x;
  float s1 = 0.0f, s2 = 0.0f;
  for (int k = blockIdx.x; k < E; k += gridDim.x) {
    const float w = mask ? mask[k] : 1.0f;
    if (w != 0.0f) {
      const float x = X[(size_t)k * HC + c];
      s1 += w * x;
      s2 += w * x * x;
    }
  }
  atomicAdd(&S1[c], s1);
  atomicAdd(&S2[c], s2);
}

__global__ void k_bn_fin(const float* __restrict__ S1, const float* __restrict__ S2,
                         const float* __restrict__ cntPtr, float cntVal,
                         float* __restrict__ mean, float* __restrict__ invstd) {
  const int c = threadIdx.x;
  const float cnt = cntPtr ? fmaxf(*cntPtr, 1.0f) : cntVal;
  const float m = S1[c] / cnt;
  mean[c] = m;
  invstd[c] = rsqrtf(S2[c] / cnt - m * m + BNEPS);
}

// bn1 apply: x_skip16 = bn(e); also n[dst[k],:] += bn(e)
__global__ void k_bn1_apply(const float* __restrict__ X, const float* __restrict__ mean,
                            const float* __restrict__ invstd, const float* __restrict__ g,
                            const float* __restrict__ be, const int* __restrict__ dst,
                            _Float16* __restrict__ out16, float* __restrict__ nmat) {
  const int k = blockIdx.x, c = threadIdx.x;
  const float v = g[c] * (X[(size_t)k * HC + c] - mean[c]) * invstd[c] + be[c];
  out16[(size_t)k * HC + c] = (_Float16)v;
  atomicAdd(&nmat[(size_t)dst[k] * HC + c], v);
}

// logits[k,:] = bn2(e2[k,:]) @ fw + fb    (ET==2, fused bn + tiny matvec)
__global__ void k_logits(const float* __restrict__ X, const float* __restrict__ mean,
                         const float* __restrict__ invstd, const float* __restrict__ g,
                         const float* __restrict__ be, const float* __restrict__ fw,
                         const float* __restrict__ fb, float* __restrict__ out) {
  __shared__ float r0[HC], r1[HC];
  const int k = blockIdx.x, c = threadIdx.x;
  const float v = g[c] * (X[(size_t)k * HC + c] - mean[c]) * invstd[c] + be[c];
  r0[c] = v * fw[c * 2 + 0];
  r1[c] = v * fw[c * 2 + 1];
  __syncthreads();
  for (int off = HC / 2; off > 0; off >>= 1) {
    if (c < off) { r0[c] += r0[c + off]; r1[c] += r1[c + off]; }
    __syncthreads();
  }
  if (c == 0) {
    out[(size_t)k * 2 + 0] = r0[0] + fb[0];
    out[(size_t)k * 2 + 1] = r1[0] + fb[1];
  }
}

// gumbel hard sample: mask = (argmax((logits+g)/tau) == 1); cnt += mask
__global__ void k_mask(const float* __restrict__ logits, const float* __restrict__ gum,
                       float* __restrict__ mask, float* __restrict__ cnt, int E) {
  const int i = blockIdx.x * blockDim.x + threadIdx.x;
  if (i >= E) return;
  const float a = logits[(size_t)i * 2 + 0] + gum[(size_t)i * 2 + 0];
  const float b = logits[(size_t)i * 2 + 1] + gum[(size_t)i * 2 + 1];
  const float m = (b > a) ? 1.0f : 0.0f;   // argmax picks index 0 on ties
  mask[i] = m;
  if (m != 0.0f) atomicAdd(cnt, 1.0f);
}

// decoder bn apply fused with masked scatter: nd[dst,:] += bn(ed)*mask
__global__ void k_bnd_scatter(const float* __restrict__ X, const float* __restrict__ mean,
                              const float* __restrict__ invstd, const float* __restrict__ g,
                              const float* __restrict__ be, const float* __restrict__ mask,
                              const int* __restrict__ dst, float* __restrict__ nd) {
  const int k = blockIdx.x;
  if (mask[k] == 0.0f) return;
  const int c = threadIdx.x;
  const float v = g[c] * (X[(size_t)k * HC + c] - mean[c]) * invstd[c] + be[c];
  atomicAdd(&nd[(size_t)dst[k] * HC + c], v);
}

// ------------------------------ orchestration ------------------------------
extern "C" void kernel_launch(void* const* d_in, const int* in_sizes, int n_in,
                              void* d_out, int out_size, void* d_ws, size_t ws_size,
                              hipStream_t stream) {
  const float* x        = (const float*)d_in[0];
  const float* gum      = (const float*)d_in[1];
  const float* enc_w1   = (const float*)d_in[2];
  const float* enc_b1   = (const float*)d_in[3];
  const float* enc_mw1  = (const float*)d_in[4];
  const float* enc_mb1  = (const float*)d_in[5];
  const float* enc_g1   = (const float*)d_in[6];
  const float* enc_be1  = (const float*)d_in[7];
  const float* enc_w2   = (const float*)d_in[8];
  const float* enc_b2   = (const float*)d_in[9];
  const float* enc_mw2  = (const float*)d_in[10];
  const float* enc_mb2  = (const float*)d_in[11];
  const float* enc_g2   = (const float*)d_in[12];
  const float* enc_be2  = (const float*)d_in[13];
  const float* enc_fw   = (const float*)d_in[14];
  const float* enc_fb   = (const float*)d_in[15];
  const float* dec_w1   = (const float*)d_in[16];
  const float* dec_b1   = (const float*)d_in[17];
  const float* dec_mw   = (const float*)d_in[18];
  const float* dec_mb   = (const float*)d_in[19];
  const float* dec_g    = (const float*)d_in[20];
  const float* dec_be   = (const float*)d_in[21];
  const float* dec_w2   = (const float*)d_in[22];
  const float* dec_b2   = (const float*)d_in[23];
  const int*   src      = (const int*)d_in[24];
  const int*   dst      = (const int*)d_in[25];
  float* out = (float*)d_out;                    // logits[E*2] ++ recon[J*NOUT]

  // --- workspace carve-out (~240 MB) ---
  char* base = (char*)d_ws; size_t off = 0;
  auto alloc = [&](size_t bytes) -> void* {
    void* p = base + off; off = (off + bytes + 255) & ~(size_t)255; return p;
  };
  float*    bufA  = (float*)alloc((size_t)EN * HC * 4);   // e -> e2 -> ed
  _Float16* bufB  = (_Float16*)alloc((size_t)EN * HC * 2);// x_skip f16
  float*    mask  = (float*)alloc((size_t)EN * 4);
  _Float16* x016  = (_Float16*)alloc((size_t)JN * NINC * 2);
  const size_t NB = (size_t)JN * HC;
  float *xw1 = (float*)alloc(NB*4), *agg1 = (float*)alloc(NB*4);
  float *P1  = (float*)alloc(NB*4), *Q1   = (float*)alloc(NB*4);
  float *nmat= (float*)alloc(NB*4), *xw2  = (float*)alloc(NB*4);
  float *agg2= (float*)alloc(NB*4), *P2   = (float*)alloc(NB*4);
  float *Q2  = (float*)alloc(NB*4), *xwd  = (float*)alloc(NB*4);
  float *aggd= (float*)alloc(NB*4), *Pd   = (float*)alloc(NB*4);
  float *Qd  = (float*)alloc(NB*4), *nd   = (float*)alloc(NB*4);
  _Float16 *h16  = (_Float16*)alloc(NB*2), *n16  = (_Float16*)alloc(NB*2);
  _Float16 *h216 = (_Float16*)alloc(NB*2), *hd16 = (_Float16*)alloc(NB*2);
  _Float16 *nd16 = (_Float16*)alloc(NB*2);
  float *xwr  = (float*)alloc((size_t)JN * NOUTC * 4);
  float *aggr = (float*)alloc((size_t)JN * NOUTC * 4);
  _Float16 *w1t  = (_Float16*)alloc((size_t)HC*NINC*2);   // enc_w1^T
  _Float16 *W1aT = (_Float16*)alloc((size_t)HC*HC*2), *W1bT = (_Float16*)alloc((size_t)HC*HC*2);
  _Float16 *w2t  = (_Float16*)alloc((size_t)HC*HC*2);
  _Float16 *W2aT = (_Float16*)alloc((size_t)HC*HC*2), *W2bT = (_Float16*)alloc((size_t)HC*HC*2);
  _Float16 *W2cT = (_Float16*)alloc((size_t)HC*HC*2);
  _Float16 *wd1t = (_Float16*)alloc((size_t)HC*NINC*2);
  _Float16 *WdaT = (_Float16*)alloc((size_t)HC*HC*2), *WdbT = (_Float16*)alloc((size_t)HC*HC*2);
  _Float16 *wd2t = (_Float16*)alloc((size_t)NOUTC*HC*2); // dec_w2^T [1024,256]
  float *S1 = (float*)alloc(HC*4), *S2 = (float*)alloc(HC*4);
  float *meanv = (float*)alloc(HC*4), *invstd = (float*)alloc(HC*4);
  float *deg = (float*)alloc(JN*4), *degd = (float*)alloc(JN*4);
  float *cnt = (float*)alloc(4);

  auto fill = [&](float* p, size_t n, float v) {
    k_fill<<<dim3(512), dim3(256), 0, stream>>>(p, n, v);
  };
  auto gemm0 = [&](const _Float16* A, const _Float16* Bt, float* C, int M, int N, int K) {
    k_gemm_wmma<<<dim3(N/64, M/16), dim3(128), 0, stream>>>(A, Bt, C, M, N, K);
  };

  // --- prep: conversions + weight transposes ---
  k_cvt_f16<<<dim3(512), dim3(256), 0, stream>>>(x, x016, (size_t)JN * NINC);
  k_wtrans<<<dim3(512), dim3(256), 0, stream>>>(enc_w1, w1t, NINC, HC, 0);
  k_wtrans<<<dim3(256), dim3(256), 0, stream>>>(enc_mw1, W1aT, HC, HC, 0);
  k_wtrans<<<dim3(256), dim3(256), 0, stream>>>(enc_mw1, W1bT, HC, HC, HC);
  k_wtrans<<<dim3(256), dim3(256), 0, stream>>>(enc_w2, w2t, HC, HC, 0);
  k_wtrans<<<dim3(256), dim3(256), 0, stream>>>(enc_mw2, W2aT, HC, HC, 0);
  k_wtrans<<<dim3(256), dim3(256), 0, stream>>>(enc_mw2, W2bT, HC, HC, HC);
  k_wtrans<<<dim3(256), dim3(256), 0, stream>>>(enc_mw2, W2cT, HC, HC, 2 * HC);
  k_wtrans<<<dim3(512), dim3(256), 0, stream>>>(dec_w1, wd1t, NINC, HC, 0);
  k_wtrans<<<dim3(256), dim3(256), 0, stream>>>(dec_mw, WdaT, HC, HC, 0);
  k_wtrans<<<dim3(256), dim3(256), 0, stream>>>(dec_mw, WdbT, HC, HC, HC);
  k_wtrans<<<dim3(512), dim3(256), 0, stream>>>(dec_w2, wd2t, HC, NOUTC, 0);
  fill(agg1, NB, 0.f); fill(nmat, NB, 0.f); fill(agg2, NB, 0.f);
  fill(aggd, NB, 0.f); fill(nd, NB, 0.f);
  fill(aggr, (size_t)JN * NOUTC, 0.f);
  fill(deg, JN, 1.f); fill(degd, JN, 1.f); fill(cnt, 1, 0.f);

  // --- encoder GCN 1: h = relu(gcn(x0, enc_w1)) ---
  gemm0(x016, w1t, xw1, JN, HC, NINC);
  k_indeg<<<dim3((EN + 255) / 256), dim3(256), 0, stream>>>(dst, nullptr, deg, EN);
  k_gcn_agg<<<dim3(EN), dim3(256), 0, stream>>>(xw1, src, dst, deg, nullptr, agg1, HC);
  k_gcn_fin_f16<<<dim3(JN), dim3(256), 0, stream>>>(agg1, xw1, deg, enc_b1, h16, HC);

  // --- edge MLP 1 (factorized) + bn1 -> x_skip (f16) + n scatter ---
  gemm0(h16, W1aT, P1, JN, HC, HC);
  gemm0(h16, W1bT, Q1, JN, HC, HC);
  k_edge_mlp<<<dim3(EN), dim3(HC), 0, stream>>>(P1, Q1, enc_mb1, src, dst, bufA);
  fill(S1, HC, 0.f); fill(S2, HC, 0.f);
  k_bn_stats<<<dim3(1024), dim3(HC), 0, stream>>>(bufA, nullptr, S1, S2, EN);
  k_bn_fin<<<dim3(1), dim3(HC), 0, stream>>>(S1, S2, nullptr, (float)EN, meanv, invstd);
  k_bn1_apply<<<dim3(EN), dim3(HC), 0, stream>>>(bufA, meanv, invstd, enc_g1, enc_be1,
                                                 dst, bufB, nmat);

  // --- encoder GCN 2 on n ---
  k_cvt_f16<<<dim3(512), dim3(256), 0, stream>>>(nmat, n16, NB);
  gemm0(n16, w2t, xw2, JN, HC, HC);
  k_gcn_agg<<<dim3(EN), dim3(256), 0, stream>>>(xw2, src, dst, deg, nullptr, agg2, HC);
  k_gcn_fin_f16<<<dim3(JN), dim3(256), 0, stream>>>(agg2, xw2, deg, enc_b2, h216, HC);

  // --- edge MLP 2: node parts + big async/WMMA GEMM with fused epilogue ---
  gemm0(h216, W2aT, P2, JN, HC, HC);
  gemm0(h216, W2bT, Q2, JN, HC, HC);
  k_gemm_edge<<<dim3(EN / 16), dim3(512), 0, stream>>>(
      bufB, W2cT, bufA, P2, Q2, enc_mb2, src, dst);

  // --- bn2 + logits + gumbel hard mask ---
  fill(S1, HC, 0.f); fill(S2, HC, 0.f);
  k_bn_stats<<<dim3(1024), dim3(HC), 0, stream>>>(bufA, nullptr, S1, S2, EN);
  k_bn_fin<<<dim3(1), dim3(HC), 0, stream>>>(S1, S2, nullptr, (float)EN, meanv, invstd);
  k_logits<<<dim3(EN), dim3(HC), 0, stream>>>(bufA, meanv, invstd, enc_g2, enc_be2,
                                              enc_fw, enc_fb, out);
  k_mask<<<dim3((EN + 255) / 256), dim3(256), 0, stream>>>(out, gum, mask, cnt, EN);
  k_indeg<<<dim3((EN + 255) / 256), dim3(256), 0, stream>>>(dst, mask, degd, EN);

  // --- decoder GCN 1 (masked) ---
  gemm0(x016, wd1t, xwd, JN, HC, NINC);
  k_gcn_agg<<<dim3(EN), dim3(256), 0, stream>>>(xwd, src, dst, degd, mask, aggd, HC);
  k_gcn_fin_f16<<<dim3(JN), dim3(256), 0, stream>>>(aggd, xwd, degd, dec_b1, hd16, HC);

  // --- decoder edge MLP (factorized) + masked bn + masked scatter to nd ---
  gemm0(hd16, WdaT, Pd, JN, HC, HC);
  gemm0(hd16, WdbT, Qd, JN, HC, HC);
  k_edge_mlp<<<dim3(EN), dim3(HC), 0, stream>>>(Pd, Qd, dec_mb, src, dst, bufA);
  fill(S1, HC, 0.f); fill(S2, HC, 0.f);
  k_bn_stats<<<dim3(1024), dim3(HC), 0, stream>>>(bufA, mask, S1, S2, EN);
  k_bn_fin<<<dim3(1), dim3(HC), 0, stream>>>(S1, S2, cnt, 0.f, meanv, invstd);
  k_bnd_scatter<<<dim3(EN), dim3(HC), 0, stream>>>(bufA, meanv, invstd, dec_g, dec_be,
                                                   mask, dst, nd);

  // --- recon = gcn(nd, dec_w2, mask) ---
  k_cvt_f16<<<dim3(512), dim3(256), 0, stream>>>(nd, nd16, NB);
  gemm0(nd16, wd2t, xwr, JN, NOUTC, HC);
  k_gcn_agg<<<dim3(EN), dim3(256), 0, stream>>>(xwr, src, dst, degd, mask, aggr, NOUTC);
  k_gcn_fin_out<<<dim3(JN), dim3(256), 0, stream>>>(aggr, xwr, degd, dec_b2,
                                                    out + (size_t)EN * 2, NOUTC);
}
